// SpatialAttention_86672440033511
// MI455X (gfx1250) — matrix operand order repaired and checked by reading, not compiled, not run on hardware
//
#include <hip/hip_runtime.h>
#include <hip/hip_bf16.h>

typedef __attribute__((ext_vector_type(16))) _Float16 v16h;
typedef __attribute__((ext_vector_type(8)))  float    v8f;
typedef __attribute__((ext_vector_type(4)))  unsigned int v4u;

#define Bx 4
#define Nx 1024
#define Cx 768
#define Hx 12
#define Dx 64

union Frag16 { v16h h; v4u u[2]; };

// A fragment (16x32 f16, MxK): lane holds row (lane%16); K halves at
// kbase + (lane/16)*8 .. +8  and  kbase + 16 + (lane/16)*8 .. +8
__device__ inline v16h load_a16(const _Float16* row, int kbase, int grp) {
    Frag16 f;
    f.u[0] = *(const v4u*)(row + kbase + grp * 8);
    f.u[1] = *(const v4u*)(row + kbase + 16 + grp * 8);
    return f.h;
}
// B fragment (32x16 f16, KxN) from a K-contiguous source row per lane:
// lane holds col (lane%16); K halves at kbase + (lane/16)*8 .. +8
// and kbase + 16 + (lane/16)*8 .. +8
__device__ inline v16h load_b16(const _Float16* row, int kbase, int grp) {
    Frag16 f;
    f.u[0] = *(const v4u*)(row + kbase + grp * 8);
    f.u[1] = *(const v4u*)(row + kbase + 16 + grp * 8);
    return f.h;
}
__device__ inline v8f wmma_f16(v16h a, v16h b, v8f c) {
    return __builtin_amdgcn_wmma_f32_16x16x32_f16(false, a, false, b, (short)0, c, false, false);
}

// ---------------- prep kernels ----------------
__global__ void cvt_f32_to_f16(const float* __restrict__ src, _Float16* __restrict__ dst, int n) {
    for (int i = blockIdx.x * blockDim.x + threadIdx.x; i < n; i += gridDim.x * blockDim.x)
        dst[i] = (_Float16)src[i];
}
// WT[out][in] = W[in][out], f32 -> f16, W is Cx x Cx
__global__ void transpose_w(const float* __restrict__ W, _Float16* __restrict__ WT) {
    int n = Cx * Cx;
    for (int idx = blockIdx.x * blockDim.x + threadIdx.x; idx < n; idx += gridDim.x * blockDim.x) {
        int o = idx / Cx, i = idx % Cx;
        WT[idx] = (_Float16)W[(size_t)i * Cx + o];
    }
}

// ---------------- QKV projection ----------------
// Block = 4 waves, all sharing one n-tile (same weight rows -> L0 hits),
// 4 adjacent m-tiles. Each wave: 16x16 tile x 3 outputs, double-buffered.
__global__ __launch_bounds__(128) void qkv_gemm(
    const _Float16* __restrict__ xh,
    const _Float16* __restrict__ WqT, const _Float16* __restrict__ WkT,
    const _Float16* __restrict__ WvT,
    const float* __restrict__ bq, const float* __restrict__ bk,
    const float* __restrict__ bv,
    _Float16* __restrict__ qh, _Float16* __restrict__ kh, _Float16* __restrict__ vT)
{
    int w    = threadIdx.x >> 5;
    int lane = threadIdx.x & 31;
    int grp  = lane >> 4;
    int col  = lane & 15;
    int nbase = blockIdx.x * 16;              // over Cx (= h*64 + d)
    int mbase = blockIdx.y * 64 + w * 16;     // over B*N

    const _Float16* arow  = xh  + (size_t)(mbase + col) * Cx;
    const _Float16* bqrow = WqT + (size_t)(nbase + col) * Cx;
    const _Float16* bkrow = WkT + (size_t)(nbase + col) * Cx;
    const _Float16* bvrow = WvT + (size_t)(nbase + col) * Cx;

    v8f aq = {}, ak = {}, av = {};
    // software pipeline: fragments for step k live while step k+1 loads fly
    v16h ca = load_a16(arow, 0, grp);
    v16h cq = load_b16(bqrow, 0, grp);
    v16h ck = load_b16(bkrow, 0, grp);
    v16h cv = load_b16(bvrow, 0, grp);
    for (int k0 = 0; k0 < Cx; k0 += 32) {
        int kn = (k0 + 32 < Cx) ? k0 + 32 : 0;   // wrap: keeps loads unconditional
        v16h na = load_a16(arow, kn, grp);
        v16h nq = load_b16(bqrow, kn, grp);
        v16h nk = load_b16(bkrow, kn, grp);
        v16h nv = load_b16(bvrow, kn, grp);
        aq = wmma_f16(ca, cq, aq);
        ak = wmma_f16(ca, ck, ak);
        av = wmma_f16(ca, cv, av);
        ca = na; cq = nq; ck = nk; cv = nv;
    }
    int b  = mbase >> 10;
    int i0 = mbase & (Nx - 1);
    int h  = nbase >> 6;
    int d  = (nbase & 63) + col;
    float vbq = bq[nbase + col], vbk = bk[nbase + col], vbv = bv[nbase + col];
    size_t bh = (size_t)b * Hx + h;
#pragma unroll
    for (int jj = 0; jj < 8; jj++) {
        int i = i0 + grp * 8 + jj;
        size_t qidx = (bh * Nx + i) * Dx + d;          // [B][H][N][D]
        qh[qidx] = (_Float16)(aq[jj] + vbq);
        kh[qidx] = (_Float16)(ak[jj] + vbk);
        size_t vidx = (bh * Dx + d) * Nx + i;          // [B][H][D][N]
        vT[vidx] = (_Float16)(av[jj] + vbv);
    }
}

// ---------------- fused attention ----------------
#define NW 8
__global__ __launch_bounds__(256) void attn_fused(
    const _Float16* __restrict__ qh, const _Float16* __restrict__ kh,
    const _Float16* __restrict__ vT, const float* __restrict__ coords,
    const float* __restrict__ Wd1, const float* __restrict__ bd1,
    const float* __restrict__ Wd2, const float* __restrict__ bd2,
    float* __restrict__ attn_out, _Float16* __restrict__ ctx)
{
    __shared__ float    ci[16][2];
    __shared__ float    wred[NW][16];
    __shared__ float    rowred[2][16];
    __shared__ _Float16 Pbuf[16][Nx + 8];        // pad 8 halves vs bank conflicts
    __shared__ float    part[NW][16][16];

    int tid  = threadIdx.x;
    int w    = tid >> 5;
    int lane = tid & 31;
    int grp  = lane >> 4;
    int col  = lane & 15;
    int ibase = blockIdx.x * 16;
    int h     = blockIdx.y;
    int b     = blockIdx.z;
    size_t bh = (size_t)b * Hx + h;

    if (tid < 32) ((float*)ci)[tid] = coords[(size_t)b * Nx * 2 + ibase * 2 + tid];
    __syncthreads();

    // distance-MLP params (uniform across block)
    float w1[16], b1v[16], w2[16];
#pragma unroll
    for (int t = 0; t < 16; t++) { w1[t] = Wd1[t]; b1v[t] = bd1[t]; w2[t] = Wd2[t * Hx + h]; }
    float b2v = bd2[h];

    float cix[8], ciy[8];
#pragma unroll
    for (int jj = 0; jj < 8; jj++) { cix[jj] = ci[grp * 8 + jj][0]; ciy[jj] = ci[grp * 8 + jj][1]; }

    // Q fragments for this i-block (K = D = 64 -> two 16x32 fragments)
    const _Float16* qrow = qh + (bh * Nx + ibase + col) * Dx;
    v16h a0 = load_a16(qrow, 0, grp);
    v16h a1 = load_a16(qrow, 32, grp);

    const float scale = 0.125f;   // 64^-0.5
    int jw = w * 128;             // this wave's column range
    v8f sc[8];

    // pipelined QK^T + bias: prefetch tile t+1's K fragments and coords
    const _Float16* kbase_p = kh + (bh * Nx + jw + col) * Dx;
    const float*    cbase_p = coords + ((size_t)b * Nx + jw + col) * 2;
    v16h kb0 = load_b16(kbase_p, 0, grp);
    v16h kb1 = load_b16(kbase_p, 32, grp);
    float cjx = cbase_p[0], cjy = cbase_p[1];
#pragma unroll
    for (int t = 0; t < 8; t++) {
        int tn = (t + 1) & 7;
        const _Float16* krow_n = kbase_p + (size_t)tn * 16 * Dx;
        v16h nb0 = load_b16(krow_n, 0, grp);
        v16h nb1 = load_b16(krow_n, 32, grp);
        float njx = cbase_p[(size_t)tn * 16 * 2 + 0];
        float njy = cbase_p[(size_t)tn * 16 * 2 + 1];

        v8f acc = {};
        acc = wmma_f16(a0, kb0, acc);
        acc = wmma_f16(a1, kb1, acc);
#pragma unroll
        for (int jj = 0; jj < 8; jj++) {
            float dx = cix[jj] - cjx, dy = ciy[jj] - cjy;
            float dist = sqrtf(fmaf(dx, dx, fmaf(dy, dy, 1e-6f)));
            float sb = b2v;
#pragma unroll
            for (int t2 = 0; t2 < 16; t2++) {
                float hx = fmaxf(fmaf(dist, w1[t2], b1v[t2]), 0.0f);
                sb = fmaf(hx, w2[t2], sb);
            }
            acc[jj] = fmaf(acc[jj], scale, sb);
        }
        sc[t] = acc;
        kb0 = nb0; kb1 = nb1; cjx = njx; cjy = njy;
    }

    // row max: within-lane over tiles, shfl over the 16-lane group, LDS over waves
    float rmax[8];
#pragma unroll
    for (int jj = 0; jj < 8; jj++) {
        float m = sc[0][jj];
#pragma unroll
        for (int t = 1; t < 8; t++) m = fmaxf(m, sc[t][jj]);
        for (int off = 8; off >= 1; off >>= 1) m = fmaxf(m, __shfl_xor(m, off, 32));
        rmax[jj] = m;
    }
    if (col == 0)
#pragma unroll
        for (int jj = 0; jj < 8; jj++) wred[w][grp * 8 + jj] = rmax[jj];
    __syncthreads();
    if (tid < 16) {
        float m = wred[0][tid];
        for (int ww = 1; ww < NW; ww++) m = fmaxf(m, wred[ww][tid]);
        rowred[0][tid] = m;
    }
    __syncthreads();
#pragma unroll
    for (int jj = 0; jj < 8; jj++) rmax[jj] = rowred[0][grp * 8 + jj];

    // exp + row sum
    float rsum[8];
#pragma unroll
    for (int jj = 0; jj < 8; jj++) rsum[jj] = 0.0f;
#pragma unroll
    for (int t = 0; t < 8; t++)
#pragma unroll
        for (int jj = 0; jj < 8; jj++) {
            float e = __expf(sc[t][jj] - rmax[jj]);
            sc[t][jj] = e;
            rsum[jj] += e;
        }
#pragma unroll
    for (int jj = 0; jj < 8; jj++) {
        float s = rsum[jj];
        for (int off = 8; off >= 1; off >>= 1) s += __shfl_xor(s, off, 32);
        rsum[jj] = s;
    }
    if (col == 0)
#pragma unroll
        for (int jj = 0; jj < 8; jj++) wred[w][grp * 8 + jj] = rsum[jj];
    __syncthreads();
    if (tid < 16) {
        float s = 0.0f;
        for (int ww = 0; ww < NW; ww++) s += wred[ww][tid];
        rowred[1][tid] = s;
    }
    __syncthreads();

    // normalize, write attn (coalesced 64B runs), stash P f16 in LDS
    float* aout = attn_out + (bh * Nx + ibase) * (size_t)Nx;
#pragma unroll
    for (int jj = 0; jj < 8; jj++) rsum[jj] = 1.0f / rowred[1][grp * 8 + jj];
#pragma unroll
    for (int t = 0; t < 8; t++) {
        int j = jw + t * 16 + col;
#pragma unroll
        for (int jj = 0; jj < 8; jj++) {
            int m = grp * 8 + jj;
            float p = sc[t][jj] * rsum[jj];
            aout[(size_t)m * Nx + j] = p;
            Pbuf[m][j] = (_Float16)p;
        }
    }
    __syncthreads();

    // P @ V: wave -> d-tile (w&3), K-half (w>>2); 16 WMMAs, double-buffered
    int dt  = (w & 3) * 16;
    int kh2 = w >> 2;
    const _Float16* vrow = vT + (bh * Dx + dt + col) * Nx + kh2 * 512;
    const _Float16* prow = &Pbuf[col][0] + kh2 * 512;
    v8f vacc = {};
    v16h pa = load_a16(prow, 0, grp);
    v16h vb = load_b16(vrow, 0, grp);
    for (int k0 = 0; k0 < 512; k0 += 32) {
        int kn = (k0 + 32 < 512) ? k0 + 32 : 0;
        v16h npa = load_a16(prow, kn, grp);
        v16h nvb = load_b16(vrow, kn, grp);
        vacc = wmma_f16(pa, vb, vacc);
        pa = npa; vb = nvb;
    }
#pragma unroll
    for (int jj = 0; jj < 8; jj++) part[w][grp * 8 + jj][col] = vacc[jj];
    __syncthreads();
    if (w < 4) {
        _Float16* crow = ctx + ((size_t)b * Nx + ibase) * Cx + h * Dx + dt;
#pragma unroll
        for (int jj = 0; jj < 8; jj++) {
            int m = grp * 8 + jj;
            float s = part[w][m][col] + part[w + 4][m][col];
            crow[(size_t)m * Cx + col] = (_Float16)s;
        }
    }
}

// ---------------- output projection ----------------
// Block = 4 waves sharing one n-tile, 4 adjacent m-tiles, double-buffered.
__global__ __launch_bounds__(128) void out_proj(
    const _Float16* __restrict__ ctx, const _Float16* __restrict__ WoT,
    const float* __restrict__ bo, float* __restrict__ out)
{
    int w    = threadIdx.x >> 5;
    int lane = threadIdx.x & 31;
    int grp  = lane >> 4;
    int col  = lane & 15;
    int nbase = blockIdx.x * 16;
    int mbase = blockIdx.y * 64 + w * 16;
    const _Float16* arow = ctx + (size_t)(mbase + col) * Cx;
    const _Float16* brow = WoT + (size_t)(nbase + col) * Cx;
    v8f acc = {};
    v16h ca = load_a16(arow, 0, grp);
    v16h cb = load_b16(brow, 0, grp);
    for (int k0 = 0; k0 < Cx; k0 += 32) {
        int kn = (k0 + 32 < Cx) ? k0 + 32 : 0;
        v16h na = load_a16(arow, kn, grp);
        v16h nb = load_b16(brow, kn, grp);
        acc = wmma_f16(ca, cb, acc);
        ca = na; cb = nb;
    }
    float bias = bo[nbase + col];
#pragma unroll
    for (int jj = 0; jj < 8; jj++)
        out[(size_t)(mbase + grp * 8 + jj) * Cx + nbase + col] = acc[jj] + bias;
}

extern "C" void kernel_launch(void* const* d_in, const int* in_sizes, int n_in,
                              void* d_out, int out_size, void* d_ws, size_t ws_size,
                              hipStream_t stream) {
    const float* x      = (const float*)d_in[0];
    const float* coords = (const float*)d_in[1];
    const float* Wq = (const float*)d_in[2];  const float* bq = (const float*)d_in[3];
    const float* Wk = (const float*)d_in[4];  const float* bk = (const float*)d_in[5];
    const float* Wv = (const float*)d_in[6];  const float* bv = (const float*)d_in[7];
    const float* Wo = (const float*)d_in[8];  const float* bo = (const float*)d_in[9];
    const float* Wd1 = (const float*)d_in[10]; const float* bd1 = (const float*)d_in[11];
    const float* Wd2 = (const float*)d_in[12]; const float* bd2 = (const float*)d_in[13];

    // workspace carve (all sizes multiples of 256B)
    char* p = (char*)d_ws;
    const size_t SZ_XH = (size_t)Bx * Nx * Cx * 2;      // 6,291,456
    const size_t SZ_W  = (size_t)Cx * Cx * 2;           // 1,179,648
    const size_t SZ_Q  = (size_t)Bx * Hx * Nx * Dx * 2; // 6,291,456
    _Float16* xh  = (_Float16*)p;               p += SZ_XH;
    _Float16* WqT = (_Float16*)p;               p += SZ_W;
    _Float16* WkT = (_Float16*)p;               p += SZ_W;
    _Float16* WvT = (_Float16*)p;               p += SZ_W;
    _Float16* WoT = (_Float16*)p;               p += SZ_W;
    _Float16* qh  = (_Float16*)p;               p += SZ_Q;
    _Float16* kh  = (_Float16*)p;               p += SZ_Q;
    _Float16* vT  = (_Float16*)p;               p += SZ_Q;
    _Float16* ctx = (_Float16*)p;               p += SZ_XH;

    float* out_ptr  = (float*)d_out;                              // [B,N,C]
    float* attn_ptr = (float*)d_out + (size_t)Bx * Nx * Cx;       // [B,H,N,N]

    // prep
    cvt_f32_to_f16<<<2048, 256, 0, stream>>>(x, xh, Bx * Nx * Cx);
    transpose_w<<<2304, 256, 0, stream>>>(Wq, WqT);
    transpose_w<<<2304, 256, 0, stream>>>(Wk, WkT);
    transpose_w<<<2304, 256, 0, stream>>>(Wv, WvT);
    transpose_w<<<2304, 256, 0, stream>>>(Wo, WoT);

    // QKV projections: 48 n-tiles x 64 m-blocks, 4 waves/block
    qkv_gemm<<<dim3(Cx / 16, (Bx * Nx) / 64), 128, 0, stream>>>(
        xh, WqT, WkT, WvT, bq, bk, bv, qh, kh, vT);

    // fused attention: (i-block, h, b)
    attn_fused<<<dim3(Nx / 16, Hx, Bx), 256, 0, stream>>>(
        qh, kh, vT, coords, Wd1, bd1, Wd2, bd2, attn_ptr, ctx);

    // output projection
    out_proj<<<dim3(Cx / 16, (Bx * Nx) / 64), 128, 0, stream>>>(ctx, WoT, bo, out_ptr);
}
